// GAT_5961414607308
// MI455X (gfx1250) — matrix operand order repaired and checked by reading, not compiled, not run on hardware
//
#include <hip/hip_runtime.h>
#include <hip/hip_bf16.h>

typedef __bf16 bf16;
typedef __attribute__((ext_vector_type(16))) __bf16 v16bf;
typedef __attribute__((ext_vector_type(8)))  __bf16 v8bf;
typedef __attribute__((ext_vector_type(8)))  float  v8f;

#define NEG_SLOPE 0.2f
#define MTILES_PER_BLOCK 8
#define WLDS_STRIDE 136   // 128 + 8 pad: 272B row stride -> 4-bank rotation, no conflicts

__device__ __forceinline__ float leaky(float v) { return v > 0.f ? v : NEG_SLOPE * v; }

// Order-preserving float<->uint mapping so atomicMax(uint) == float max.
__device__ __forceinline__ unsigned enc_f(float f) {
    unsigned u = __float_as_uint(f);
    return (u & 0x80000000u) ? ~u : (u | 0x80000000u);
}
__device__ __forceinline__ float dec_f(unsigned k) {
    unsigned u = (k & 0x80000000u) ? (k & 0x7FFFFFFFu) : ~k;
    return __uint_as_float(u);
}

// ---------------------------------------------------------------------------
// h = X[N,128] @ W[128,128], f32 in, bf16 WMMA, f32 out.
// Block = 8 waves (32x8); wave w owns column tile w (16 cols) and iterates
// MTILES_PER_BLOCK row tiles. W^T is staged to LDS as bf16 once per block;
// each wave loads its 4 B fragments (two ds_load_b128 each) once, then the
// M loop is pure: A gather+cvt -> 4x v_wmma -> C store.
// ---------------------------------------------------------------------------
__global__ __launch_bounds__(256) void gat_gemm_bf16_wmma(
    const float* __restrict__ X, const float* __restrict__ W,
    float* __restrict__ Hout, int N)
{
    __shared__ bf16 wldsT[128 * WLDS_STRIDE];  // [col][k], padded

    // Stage W^T: coalesced global reads, strided (conflict-light) LDS stores.
    int t = threadIdx.y * 32 + threadIdx.x;
    for (int idx = t; idx < 128 * 128; idx += 256) {
        int k = idx >> 7, col = idx & 127;
        wldsT[col * WLDS_STRIDE + k] = (bf16)W[idx];
    }
    __syncthreads();

    const int lane  = threadIdx.x;
    const int sel   = lane >> 4;      // 0: lanes 0-15, 1: lanes 16-31
    const int l15   = lane & 15;
    const int tileN = threadIdx.y;    // 0..7 (covers all 128 output cols)
    const int col   = tileN * 16 + l15;

    // B fragments: lanes 0-15 hold K=k0..k0+15, lanes 16-31 K=k0+16..k0+31,
    // at column `col`. 16 consecutive K are contiguous in wldsT -> 2x b128.
    v16bf bfrag[4];
    const bf16* bcol = wldsT + col * WLDS_STRIDE;
#pragma unroll
    for (int kk = 0; kk < 4; ++kk) {
        const int kb = kk * 32 + sel * 16;
        v8bf b0 = *(const v8bf*)(bcol + kb);
        v8bf b1 = *(const v8bf*)(bcol + kb + 8);
#pragma unroll
        for (int i = 0; i < 8; ++i) { bfrag[kk][i] = b0[i]; bfrag[kk][8 + i] = b1[i]; }
    }

    const int tiles = N >> 4;
    const int tmEnd0 = blockIdx.x * MTILES_PER_BLOCK + MTILES_PER_BLOCK;
    const int tmEnd  = tmEnd0 < tiles ? tmEnd0 : tiles;

    for (int tm = blockIdx.x * MTILES_PER_BLOCK; tm < tmEnd; ++tm) {
        const int row = tm * 16 + l15;
        const float* xrow = X + (size_t)row * 128;
        v8f acc = {};
#pragma unroll
        for (int k0 = 0; k0 < 128; k0 += 32) {
            // A fragment: contiguous K order, same as bfrag.
            // lanes 0-15: a[0..7]=K k0..+7,    a[8..15]=K k0+8..+15
            // lanes16-31: a[0..7]=K k0+16..+23, a[8..15]=K k0+24..+31
            const float4* p0 = (const float4*)(xrow + k0 + sel * 16);
            const float4* p1 = (const float4*)(xrow + k0 + sel * 16 + 8);
            float4 f0 = p0[0], f1 = p0[1], f2 = p1[0], f3 = p1[1];
            v16bf a;
            a[0]=(bf16)f0.x;  a[1]=(bf16)f0.y;  a[2]=(bf16)f0.z;  a[3]=(bf16)f0.w;
            a[4]=(bf16)f1.x;  a[5]=(bf16)f1.y;  a[6]=(bf16)f1.z;  a[7]=(bf16)f1.w;
            a[8]=(bf16)f2.x;  a[9]=(bf16)f2.y;  a[10]=(bf16)f2.z; a[11]=(bf16)f2.w;
            a[12]=(bf16)f3.x; a[13]=(bf16)f3.y; a[14]=(bf16)f3.z; a[15]=(bf16)f3.w;

            acc = __builtin_amdgcn_wmma_f32_16x16x32_bf16(
                      false, a, false, bfrag[k0 >> 5], (short)0, acc, false, false);
        }
        // C layout: VGPR r -> M = r + sel*8, N = l15.
        float* obase = Hout + ((size_t)tm * 16 + sel * 8) * 128 + tileN * 16 + l15;
#pragma unroll
        for (int r = 0; r < 8; ++r) obase[(size_t)r * 128] = acc[r];
    }
}

// ---------------------------------------------------------------------------
// al_src[n,h], al_dst[n,h]: per-(node,head) dot over C=64 channels.
// ---------------------------------------------------------------------------
__global__ void gat_node_alpha(const float* __restrict__ Hf,
                               const float* __restrict__ a_src,
                               const float* __restrict__ a_dst,
                               float* __restrict__ al_s, float* __restrict__ al_d,
                               int NH)
{
    int i = blockIdx.x * blockDim.x + threadIdx.x;
    if (i >= NH) return;
    int n = i >> 1, hh = i & 1;
    const float4* hp = (const float4*)(Hf + (size_t)n * 128 + hh * 64);
    const float4* as = (const float4*)(a_src + hh * 64);
    const float4* ad = (const float4*)(a_dst + hh * 64);
    float s = 0.f, d = 0.f;
#pragma unroll 4
    for (int c = 0; c < 16; ++c) {
        float4 v = hp[c], a = as[c], b = ad[c];
        s += v.x * a.x + v.y * a.y + v.z * a.z + v.w * a.w;
        d += v.x * b.x + v.y * b.y + v.z * b.z + v.w * b.w;
    }
    al_s[i] = s;
    al_d[i] = d;
}

// ---------------------------------------------------------------------------
// Edge pass 1: e = leaky(al_src[src] + al_dst[dst]); segment max via ordered
// uint atomics. Self-loop edges appended implicitly at e >= E.
// ---------------------------------------------------------------------------
__global__ void gat_edge_pass1(const long long* __restrict__ ei, int E, int Etot,
                               const float* __restrict__ al_s,
                               const float* __restrict__ al_d,
                               float* __restrict__ ev, unsigned* __restrict__ menc)
{
    int e = blockIdx.x * blockDim.x + threadIdx.x;
    if (e >= Etot) return;
    int s, d;
    if (e < E) { s = (int)ei[e]; d = (int)ei[(size_t)E + e]; }
    else       { s = e - E; d = s; }
#pragma unroll
    for (int hh = 0; hh < 2; ++hh) {
        float v = leaky(al_s[s * 2 + hh] + al_d[d * 2 + hh]);
        ev[(size_t)e * 2 + hh] = v;
        atomicMax(&menc[d * 2 + hh], enc_f(v));
    }
}

// Edge pass 2: ex = exp(e - m[dst]); segment sum of ex into denom.
__global__ void gat_edge_pass2(const long long* __restrict__ ei, int E, int Etot,
                               const unsigned* __restrict__ menc,
                               float* __restrict__ ev, float* __restrict__ denom)
{
    int e = blockIdx.x * blockDim.x + threadIdx.x;
    if (e >= Etot) return;
    int d;
    if (e < E) { d = (int)ei[(size_t)E + e]; }
    else       { d = e - E; }
#pragma unroll
    for (int hh = 0; hh < 2; ++hh) {
        float mx = dec_f(menc[d * 2 + hh]);
        float ex = __expf(ev[(size_t)e * 2 + hh] - mx);
        ev[(size_t)e * 2 + hh] = ex;
        atomicAdd(&denom[d * 2 + hh], ex);
    }
}

// Edge pass 3: out[dst] += alpha * h[src]. One wave32 per edge; each lane owns
// 4 contiguous channels (float4 gather, 4 f32 atomics) -> coalesced 512B/wave.
__global__ __launch_bounds__(256) void gat_edge_pass3(
    const long long* __restrict__ ei, int E, int Etot,
    const float* __restrict__ Hf, const float* __restrict__ ev,
    const float* __restrict__ denom, float* __restrict__ Out)
{
    int g = blockIdx.x * 8 + (threadIdx.x >> 5);
    if (g >= Etot) return;
    int lane = threadIdx.x & 31;
    int s, d;
    if (g < E) { s = (int)ei[g]; d = (int)ei[(size_t)E + g]; }
    else       { s = g - E; d = s; }
    int c0 = lane * 4;            // channels c0..c0+3, head = c0>>6
    int hh = c0 >> 6;
    float alpha = ev[(size_t)g * 2 + hh] / denom[d * 2 + hh];
    float4 hv = *(const float4*)(Hf + (size_t)s * 128 + c0);
    float* op = Out + (size_t)d * 128 + c0;
    atomicAdd(op + 0, alpha * hv.x);
    atomicAdd(op + 1, alpha * hv.y);
    atomicAdd(op + 2, alpha * hv.z);
    atomicAdd(op + 3, alpha * hv.w);
}

// Epilogue: Act = leaky((residual ? Act : 0) + Out + bias). In-place safe.
__global__ void gat_epilogue(const float* __restrict__ Out,
                             const float* __restrict__ bias,
                             float* __restrict__ Act, int total, int residual)
{
    int i = blockIdx.x * blockDim.x + threadIdx.x;
    if (i >= total) return;
    float v = Out[i] + bias[i & 127];
    if (residual) v += Act[i];
    Act[i] = leaky(v);
}

// logits[n] = dot(Act[n,:], lin_w) + lin_b. One wave32 per node, shuffle tree.
__global__ __launch_bounds__(256) void gat_final_linear(
    const float* __restrict__ Act, const float* __restrict__ lw,
    const float* __restrict__ lb, float* __restrict__ out, int N)
{
    int n = blockIdx.x * 8 + (threadIdx.x >> 5);
    if (n >= N) return;
    int lane = threadIdx.x & 31;
    float4 a = *(const float4*)(Act + (size_t)n * 128 + lane * 4);
    float4 w = *(const float4*)(lw + lane * 4);
    float s = a.x * w.x + a.y * w.y + a.z * w.z + a.w * w.w;
#pragma unroll
    for (int off = 16; off; off >>= 1) s += __shfl_down(s, off, 32);
    if (lane == 0) out[n] = s + lb[0];
}

extern "C" void kernel_launch(void* const* d_in, const int* in_sizes, int n_in,
                              void* d_out, int out_size, void* d_ws, size_t ws_size,
                              hipStream_t stream)
{
    (void)n_in; (void)out_size; (void)ws_size;
    const float*     x  = (const float*)d_in[0];
    const long long* ei = (const long long*)d_in[1];   // int64 [2,E]
    const int N    = in_sizes[0] / 128;
    const int E    = in_sizes[1] / 2;
    const int Etot = E + N;                             // + self loops
    const int H    = 2;

    // Workspace carve-out (floats). Total ~21.3M floats (~85 MB).
    float* ws = (float*)d_ws;
    const size_t NF = (size_t)N * 128;
    float*    bufA  = ws;                                // activations (layer in/out)
    float*    bufH  = bufA + NF;                         // h = X@W
    float*    bufO  = bufH + NF;                         // attention accumulation
    float*    al_s  = bufO + NF;
    float*    al_d  = al_s + (size_t)N * H;
    unsigned* menc  = (unsigned*)(al_d + (size_t)N * H); // segment max (encoded)
    float*    denom = (float*)(menc + (size_t)N * H);
    float*    ev    = denom + (size_t)N * H;             // per-edge logits/exp

    dim3 gemmBlock(32, 8);
    const int tiles    = N / 16;                               // 3125 for N=50000
    const int gemmGrid = (tiles + MTILES_PER_BLOCK - 1) / MTILES_PER_BLOCK;
    const int nhBlocks = (N * H + 255) / 256;
    const int eBlocks  = (Etot + 255) / 256;
    const int e3Blocks = (Etot + 7) / 8;
    const int epBlocks = (int)((NF + 255) / 256);
    const int fiBlocks = (N + 7) / 8;

    for (int layer = 0; layer < 3; ++layer) {
        const float* W    = (const float*)d_in[2 + 4 * layer];
        const float* asrc = (const float*)d_in[3 + 4 * layer];
        const float* adst = (const float*)d_in[4 + 4 * layer];
        const float* bias = (const float*)d_in[5 + 4 * layer];
        const float* Xin  = (layer == 0) ? x : bufA;

        gat_gemm_bf16_wmma<<<gemmGrid, gemmBlock, 0, stream>>>(Xin, W, bufH, N);
        gat_node_alpha<<<nhBlocks, 256, 0, stream>>>(bufH, asrc, adst, al_s, al_d, N * H);

        hipMemsetAsync(menc,  0, (size_t)N * H * sizeof(unsigned), stream);
        hipMemsetAsync(denom, 0, (size_t)N * H * sizeof(float), stream);
        hipMemsetAsync(bufO,  0, NF * sizeof(float), stream);

        gat_edge_pass1<<<eBlocks, 256, 0, stream>>>(ei, E, Etot, al_s, al_d, ev, menc);
        gat_edge_pass2<<<eBlocks, 256, 0, stream>>>(ei, E, Etot, menc, ev, denom);
        gat_edge_pass3<<<e3Blocks, 256, 0, stream>>>(ei, E, Etot, bufH, ev, denom, bufO);
        gat_epilogue<<<epBlocks, 256, 0, stream>>>(bufO, bias, bufA, (int)NF, layer > 0 ? 1 : 0);
    }

    gat_final_linear<<<fiBlocks, 256, 0, stream>>>(
        bufA, (const float*)d_in[14], (const float*)d_in[15], (float*)d_out, N);
}
